// Model_16028817949145
// MI455X (gfx1250) — matrix operand order repaired and checked by reading, not compiled, not run on hardware
//
#include <hip/hip_runtime.h>

// Problem constants (match reference)
constexpr int B_ = 8, N_ = 2000, D_ = 480, L_ = 8, K_ = 4096, T1_ = 6;
constexpr int TN = 48;            // workgroup N-tile: 3 x 16, 480 = 10 * 48 exactly
constexpr int NSUB = TN / 16;     // 3 WMMA subtiles, compile-time

typedef __attribute__((ext_vector_type(16))) __bf16          bf16x16;
typedef __attribute__((ext_vector_type(8)))  float           f32x8;
typedef __attribute__((ext_vector_type(8)))  unsigned short  u16x8;
typedef __attribute__((ext_vector_type(16))) unsigned short  u16x16;
typedef __attribute__((ext_vector_type(4)))  unsigned int    u32x4;
typedef __attribute__((ext_vector_type(8)))  int             i32x8;
typedef __attribute__((ext_vector_type(4)))  int             i32x4;

#if defined(__has_builtin)
#if __has_builtin(__builtin_amdgcn_tensor_load_to_lds)
#define HAS_TDM 1
#endif
#endif

__device__ __forceinline__ unsigned short f2bf(float f) {
    unsigned u = __float_as_uint(f);
    u += 0x7FFFu + ((u >> 16) & 1u);          // round-to-nearest-even
    return (unsigned short)(u >> 16);
}

// ---------------------------------------------------------------------------
// 1) zero output + counts (must run every call: we accumulate into d_out)
__global__ void k_zero(float* __restrict__ out, int out_size, float* __restrict__ counts) {
    int i = blockIdx.x * blockDim.x + threadIdx.x;
    if (i < out_size)                 out[i] = 0.f;
    else if (i < out_size + B_ * N_)  counts[i - out_size] = 0.f;
}

// 2) W_proj fp32 -> bf16 (row-major copy; rows of W_proj are K-contiguous
//    columns of B = W_proj^T, which is exactly the WMMA B-fragment layout)
__global__ void k_wconv(const float* __restrict__ W, unsigned short* __restrict__ Wbf) {
    int i = blockIdx.x * blockDim.x + threadIdx.x;
    if (i < D_ * D_) Wbf[i] = f2bf(W[i]);
}

// 3) leaf counts per segment id
__global__ void k_counts(const int* __restrict__ seg, const unsigned char* __restrict__ is_end,
                         float* __restrict__ counts) {
    int i = blockIdx.x * blockDim.x + threadIdx.x;   // over B*L*K
    if (i >= B_ * L_ * K_) return;
    if (is_end[i]) {
        int b = i / (L_ * K_);
        atomicAdd(&counts[b * N_ + seg[i]], 1.0f);
    }
}

// 4) level 0: z0 = feats[seg0] (bf16 state) + end-node scatter
__global__ void k_level0(const float* __restrict__ feats, const int* __restrict__ seg,
                         const unsigned char* __restrict__ is_end,
                         unsigned short* __restrict__ z0, float* __restrict__ zend) {
    const int CH = D_ / 4;
    long i = (long)blockIdx.x * blockDim.x + threadIdx.x;  // B*K*(D/4)
    if (i >= (long)B_ * K_ * CH) return;
    int  c4   = (int)(i % CH);
    long bk   = i / CH;
    int  node = (int)(bk % K_);
    int  b    = (int)(bk / K_);
    long sidx = ((long)b * L_) * K_ + node;                // level 0
    int  s    = seg[sidx];
    float4 v  = *(const float4*)(feats + ((long)b * N_ + s) * D_ + c4 * 4);
    unsigned short* zp = z0 + ((long)b * K_ + node) * D_ + c4 * 4;
    zp[0] = f2bf(v.x); zp[1] = f2bf(v.y); zp[2] = f2bf(v.z); zp[3] = f2bf(v.w);
    if (is_end[sidx]) {
        float* d = zend + ((long)b * N_ + s) * D_ + c4 * 4;
        atomicAdd(d + 0, v.x); atomicAdd(d + 1, v.y);
        atomicAdd(d + 2, v.z); atomicAdd(d + 3, v.w);
    }
}

// 5) one scan level: z_l = alpha * (z_{l-1}[parent] @ W_proj^T) + h_l
//    WMMA bf16 16x16x32, f32 accumulate. Workgroup = 8 waves = 256x48 tile;
//    each wave owns two 16-row A tiles that share every B fragment (halves
//    LDS reads per WMMA, two independent WMMAs per ds_load pair).
//    W tile (48 cols x 480 K = 46KB) staged in LDS once via the Tensor Data
//    Mover (TENSORcnt) when available, else cooperative vector copy.
__global__ __launch_bounds__(256)
void k_level(const unsigned short* __restrict__ zprev,
             unsigned short* __restrict__ znext,
             const unsigned short* __restrict__ Wbf,
             const float* __restrict__ feats,
             const float* __restrict__ attn,
             const int* __restrict__ seg,
             const int* __restrict__ parent,
             const unsigned char* __restrict__ is_end,
             float* __restrict__ zend, int l) {
    __shared__ __align__(32) unsigned short sW[TN * D_];   // [col][K], 960B per col

    const int lane   = threadIdx.x & 31;
    const int wave   = threadIdx.x >> 5;
    const int b      = blockIdx.z;
    const int m_base = blockIdx.x * 256 + wave * 32;       // 2 x 16 rows per wave
    const int n_base = blockIdx.y * TN;                    // 3 x 16-col subtiles
    const int half   = lane >> 4;
    const int col_l  = lane & 15;

#ifdef HAS_TDM
    // TDM: one 2D descriptor moves the whole 48 x 480 bf16 tile (48 rows of
    // 240 dwords, contiguous in both global and LDS). Issued by wave 0 only;
    // completion on TENSORcnt, then workgroup barrier publishes LDS.
    if (wave == 0) {
        unsigned           lds = (unsigned)(unsigned long long)(void*)sW;
        unsigned long long ga  = (unsigned long long)(const void*)(Wbf + (long)n_base * D_);
        u32x4 g0;
        g0[0] = 1u;                                        // count=1, no gather
        g0[1] = lds;                                       // lds_addr
        g0[2] = (unsigned)ga;                              // global_addr[31:0]
        g0[3] = (unsigned)((ga >> 32) & 0x1FFFFFFu) | (2u << 30);  // addr[56:32] | type=2
        i32x8 g1;
        g1[0] = 0x00020000;          // workgroup_mask=0, data_size=2 (4B)
        g1[1] = 240 << 16;           // tensor_dim0 = 240 dwords
        g1[2] = 48 << 16;            // tensor_dim1 = 48 rows
        g1[3] = 240 << 16;           // tile_dim0 = 240
        g1[4] = 48;                  // tile_dim1 = 48, tile_dim2 = 0
        g1[5] = 240;                 // tensor_dim0_stride = 240 dwords
        g1[6] = 0;
        g1[7] = 0;                   // tensor_dim1_stride = 0 (2D)
        i32x4 gz = {0, 0, 0, 0};
#if __clang_major__ >= 23
        i32x8 gz8 = {};
        __builtin_amdgcn_tensor_load_to_lds(g0, g1, gz, gz, gz8, 0);
#else
        __builtin_amdgcn_tensor_load_to_lds(g0, g1, gz, gz, 0);
#endif
        __builtin_amdgcn_s_wait_tensorcnt(0);
    }
    __syncthreads();
#else
    // Fallback: cooperative stage, u16x8 (16B) chunks; 48*60 over 256 threads.
    constexpr int CPC = D_ / 8;                            // 60 chunks per column
    for (int i = threadIdx.x; i < TN * CPC; i += 256) {
        int col   = i / CPC;
        int chunk = i % CPC;
        ((u16x8*)sW)[i] = *(const u16x8*)(Wbf + (long)(n_base + col) * D_ + chunk * 8);
    }
    __syncthreads();
#endif

    const long segL  = ((long)b * L_ + l) * K_;
    const long segLm = segL - K_;

    // A fragment sources: lane holds M row = lane%16 of each 16-row tile,
    // parent-gathered bf16 rows.
    const int par0 = parent[segL + m_base + col_l];
    const int par1 = parent[segL + m_base + 16 + col_l];
    const unsigned short* arow0 = zprev + ((long)b * K_ + par0) * D_;
    const unsigned short* arow1 = zprev + ((long)b * K_ + par1) * D_;

    f32x8 acc0[NSUB] = {};
    f32x8 acc1[NSUB] = {};
    union AU { u16x8  h[2]; bf16x16 v; };
    union BU { u16x16 u;    bf16x16 v; };

    for (int kk = 0; kk < D_ / 32; ++kk) {
        const int k0 = kk * 32;
        AU a0, a1;
        // A 16-bit layout: elems 0..7 -> K = half*8 + j ; elems 8..15 -> K = 16 + half*8 + j
        a0.h[0] = *(const u16x8*)(arow0 + k0 + half * 8);
        a0.h[1] = *(const u16x8*)(arow0 + k0 + 16 + half * 8);
        a1.h[0] = *(const u16x8*)(arow1 + k0 + half * 8);
        a1.h[1] = *(const u16x8*)(arow1 + k0 + 16 + half * 8);
#pragma unroll
        for (int ns = 0; ns < NSUB; ++ns) {
            BU bm;
            // B 16-bit layout: elems 0..7 -> K = half*8 + j ; elems 8..15 -> K = 16 + half*8 + j, col = lane%16 (from LDS)
            bm.u.lo = *(const u16x8*)(sW + (ns * 16 + col_l) * D_ + k0 + half * 8);
            bm.u.hi = *(const u16x8*)(sW + (ns * 16 + col_l) * D_ + k0 + 16 + half * 8);
            acc0[ns] = __builtin_amdgcn_wmma_f32_16x16x32_bf16(
                false, a0.v, false, bm.v, (short)0, acc0[ns], false, false);
            acc1[ns] = __builtin_amdgcn_wmma_f32_16x16x32_bf16(
                false, a1.v, false, bm.v, (short)0, acc1[ns], false, false);
        }
    }

    // Epilogue. C/D layout: VGPR r holds M = half*8 + r, N = lane%16.
    auto epilogue = [&](const f32x8 (&acc)[NSUB], int mrow_base) {
#pragma unroll
        for (int r = 0; r < 8; ++r) {
            const int   node  = mrow_base + half * 8 + r;
            const int   s     = seg[segL + node];
            const int   par   = parent[segL + node];
            const int   ps    = seg[segLm + par];
            const float alpha = attn[((long)b * N_ + ps) * N_ + s];
            const bool  endn  = is_end[segL + node] != 0;
            const float* hrow = feats + ((long)b * N_ + s) * D_;
            unsigned short* zr = znext + ((long)b * K_ + node) * D_;
            float* er = zend + ((long)b * N_ + s) * D_;
#pragma unroll
            for (int ns = 0; ns < NSUB; ++ns) {
                const int   col = n_base + ns * 16 + col_l;
                const float val = alpha * acc[ns][r] + hrow[col];
                zr[col] = f2bf(val);
                if (endn) atomicAdd(er + col, val);
            }
        }
    };
    epilogue(acc0, m_base);
    epilogue(acc1, m_base + 16);
}

// 6) normalize leaf means, predictor, mask
__global__ void k_finalize(float* __restrict__ zend, const float* __restrict__ counts,
                           const float* __restrict__ Wpred, const float* __restrict__ bpred,
                           float* __restrict__ preds, float* __restrict__ maskout) {
    int i = blockIdx.x * blockDim.x + threadIdx.x;     // over B*N
    if (i >= B_ * N_) return;
    float c    = counts[i];
    float mask = c > 0.f ? 1.f : 0.f;
    float inv  = 1.f / fmaxf(c, 1.f);
    float* z = zend + (long)i * D_;
    float p[T1_];
#pragma unroll
    for (int t = 0; t < T1_; ++t) p[t] = 0.f;
    for (int d = 0; d < D_; ++d) {
        float v = z[d] * inv;
        z[d] = v;
#pragma unroll
        for (int t = 0; t < T1_; ++t) p[t] += v * Wpred[t * D_ + d];
    }
#pragma unroll
    for (int t = 0; t < T1_; ++t) preds[(long)i * T1_ + t] = mask * (p[t] + bpred[t]);
    maskout[i] = mask;
}

// ---------------------------------------------------------------------------
extern "C" void kernel_launch(void* const* d_in, const int* in_sizes, int n_in,
                              void* d_out, int out_size, void* d_ws, size_t ws_size,
                              hipStream_t stream) {
    (void)in_sizes; (void)n_in; (void)ws_size;
    const float*         feats  = (const float*)d_in[0];   // [B,N,NH,C] = [B,N,D]
    const float*         attn   = (const float*)d_in[1];   // [B,N,N]
    const int*           seg    = (const int*)d_in[2];     // [B,L,K]
    const int*           parent = (const int*)d_in[3];     // [B,L,K]
    const unsigned char* is_end = (const unsigned char*)d_in[4];  // bool [B,L,K]
    const float*         W_proj = (const float*)d_in[5];   // [D,D]
    const float*         W_pred = (const float*)d_in[6];   // [T1,D]
    const float*         b_pred = (const float*)d_in[7];   // [T1]

    // ws layout: Wbf (0.46MB) | zbuf0 (30MB) | zbuf1 (30MB) | counts (64KB) ~= 60.5MB
    char* ws = (char*)d_ws;
    const size_t WBF_BYTES = (size_t)D_ * D_ * 2;          // 460800 (32B aligned)
    const size_t Z_BYTES   = (size_t)B_ * K_ * D_ * 2;     // 31457280
    unsigned short* Wbf    = (unsigned short*)ws;
    unsigned short* z0     = (unsigned short*)(ws + WBF_BYTES);
    unsigned short* z1     = (unsigned short*)(ws + WBF_BYTES + Z_BYTES);
    float*          counts = (float*)(ws + WBF_BYTES + 2 * Z_BYTES);

    float* preds   = (float*)d_out;                         // [B,N,T1]
    float* zend    = preds + (long)B_ * N_ * T1_;           // [B,N,D]
    float* maskout = zend + (long)B_ * N_ * D_;             // [B,N]

    {   // zero outputs + counts (determinism across graph replays)
        int tot = out_size + B_ * N_;
        k_zero<<<(tot + 255) / 256, 256, 0, stream>>>((float*)d_out, out_size, counts);
    }
    k_wconv<<<(D_ * D_ + 255) / 256, 256, 0, stream>>>(W_proj, Wbf);
    k_counts<<<(B_ * L_ * K_ + 255) / 256, 256, 0, stream>>>(seg, is_end, counts);
    {
        long tot = (long)B_ * K_ * (D_ / 4);
        k_level0<<<(int)((tot + 255) / 256), 256, 0, stream>>>(feats, seg, is_end, z0, zend);
    }
    unsigned short* zp = z0;
    unsigned short* zn = z1;
    dim3 grid(K_ / 256, D_ / TN, B_);                       // (16, 10, 8)
    for (int l = 1; l < L_; ++l) {   // sequential scan: launch = device-wide barrier
        k_level<<<grid, 256, 0, stream>>>(zp, zn, Wbf, feats, attn, seg, parent,
                                          is_end, zend, l);
        unsigned short* t = zp; zp = zn; zn = t;
    }
    k_finalize<<<(B_ * N_ + 255) / 256, 256, 0, stream>>>(zend, counts, W_pred, b_pred,
                                                          preds, maskout);
}
